// DenseSE3_51926154608783
// MI455X (gfx1250) — compile-verified, matrix operand order checked by reading, not run on hardware
//
#include <hip/hip_runtime.h>
#include <math.h>

#define B_ 2
#define C_ 16
#define H_ 32
#define W_ 32
#define N_ (H_*W_)      // 1024
#define NT 64           // 16-wide tiles per matrix dim

typedef __attribute__((ext_vector_type(2)))  float    v2f;
typedef __attribute__((ext_vector_type(8)))  float    v8f;
typedef __attribute__((ext_vector_type(16))) _Float16 v16h;

// ---------------------------------------------------------------------------
// Kernel 0: per-point precompute.
// pd[idx*8] = { X, Y, z, tgt_u, tgt_v, tgt_invz, ||e||^2, 0 }
// where (X,Y,z) = unprojected point, tgt = pie(T_k x_k) + revision_k.
// ---------------------------------------------------------------------------
__global__ void k_point(const float* __restrict__ emb, const float* __restrict__ rev,
                        const float* __restrict__ depth, const float* __restrict__ pix,
                        const float* __restrict__ Tmat, float* __restrict__ pd)
{
    int idx = blockIdx.x * blockDim.x + threadIdx.x;
    if (idx >= B_ * N_) return;
    int b = idx / N_, n = idx % N_;
    float fx = pix[b*16+0], fy = pix[b*16+5], x0 = pix[b*16+2], y0 = pix[b*16+6];
    float u = (float)(n % W_), v = (float)(n / W_);
    float z = depth[b*N_ + n];
    float X = (u - x0) * z / fx;
    float Y = (v - y0) * z / fy;
    const float* T = Tmat + (size_t)(b*N_ + n)*16;
    float Px = T[0]*X + T[1]*Y + T[2]*z  + T[3];
    float Py = T[4]*X + T[5]*Y + T[6]*z  + T[7];
    float Pz = T[8]*X + T[9]*Y + T[10]*z + T[11];
    float invz = 1.0f / Pz;
    float uu = fx*Px*invz + x0;
    float vv = fy*Py*invz + y0;
    float s = 0.f;
    #pragma unroll
    for (int c = 0; c < C_; ++c) { float e = emb[(b*C_ + c)*N_ + n]; s += e*e; }
    float* o = pd + (size_t)idx*8;
    o[0] = X;  o[1] = Y;  o[2] = z;
    o[3] = uu   + rev[(b*3+0)*N_ + n];
    o[4] = vv   + rev[(b*3+1)*N_ + n];
    o[5] = invz + rev[(b*3+2)*N_ + n];
    o[6] = s;   o[7] = 0.f;
}

// ---------------------------------------------------------------------------
// Kernel A: affinity matrix via WMMA Gram tiles.
// One wave32 computes a 16x16 tile of G = E E^T (K = C = 16), then
// aff = exp(-sqrt(max(0, s_i + s_k - 2 G))).
// ---------------------------------------------------------------------------
__global__ void k_aff(const float* __restrict__ emb, const float* __restrict__ pd,
                      float* __restrict__ aff)
{
    int gtid = blockIdx.x * blockDim.x + threadIdx.x;
    int wave = gtid >> 5;
    int lane = gtid & 31;
    if (wave >= B_ * NT * NT) return;
    int b  = wave / (NT*NT);
    int t  = wave % (NT*NT);
    int iT = t / NT, kT = t % NT;
    int h = lane >> 4;        // half-wave
    int m = lane & 15;        // A row / B col within tile
    const float* eb = emb + (size_t)b*C_*N_;

    v8f acc = {};
#if __has_builtin(__builtin_amdgcn_wmma_f32_16x16x4_f32)
    // 32-bit A 16x4: lanes0-15 hold K=2h.., VGPR0->K=2h, VGPR1->K=2h+1; B symmetric.
    #pragma unroll
    for (int c4 = 0; c4 < 4; ++c4) {
        int k0 = 4*c4 + 2*h;
        v2f av, bv;
        av.x = eb[(size_t)(k0    )*N_ + iT*16 + m];
        av.y = eb[(size_t)(k0 + 1)*N_ + iT*16 + m];
        bv.x = eb[(size_t)(k0    )*N_ + kT*16 + m];
        bv.y = eb[(size_t)(k0 + 1)*N_ + kT*16 + m];
        acc = __builtin_amdgcn_wmma_f32_16x16x4_f32(
            false, av, false, bv, (short)0, acc, false, false);
    }
#else
    // Fallback: f16 16x16x32 WMMA, K padded 16->32 with zeros (probe-confirmed builtin).
    v16h av, bv;
    #pragma unroll
    for (int j = 0; j < 16; ++j) { av[j] = (_Float16)0.f; bv[j] = (_Float16)0.f; }
    #pragma unroll
    for (int j = 0; j < 8; ++j)   // A: lanes0-15 K=0..7 / K=16..23(zero); lanes16-31 K=8..15
        av[j] = (_Float16)eb[(size_t)(8*h + j)*N_ + iT*16 + m];
    if (h == 0) {                  // B: lanes0-15 hold ch0..15 at K=0..7,16..23; lanes16-31 (K=8..15,24..31) zero
        #pragma unroll
        for (int j = 0; j < 16; ++j)
            bv[j] = (_Float16)eb[(size_t)j*N_ + kT*16 + m];
    }
    acc = __builtin_amdgcn_wmma_f32_16x16x32_f16(
        false, av, false, bv, (short)0, acc, false, false);
#endif

    float sk = pd[(size_t)(b*N_ + kT*16 + m)*8 + 6];
    float* ab = aff + (size_t)b*N_*N_;
    #pragma unroll
    for (int vv = 0; vv < 8; ++vv) {
        int i = iT*16 + vv + 8*h;               // C/D layout: row = vgpr + 8*(lane/16)
        float si = pd[(size_t)(b*N_ + i)*8 + 6];
        float d2 = fmaxf(si + sk - 2.f*acc[vv], 0.f);
        ab[(size_t)i*N_ + kT*16 + m] = expf(-sqrtf(d2));
    }
}

// ---------------------------------------------------------------------------
// Kernel B: one block per output point i. 256 threads reduce over k,
// accumulate symmetric 6x6 H (21) + rhs (6), wave32 shuffle reduce,
// lane0 solves 6x6, expmap, composes dT @ Tmat_i.
// ---------------------------------------------------------------------------
__global__ __launch_bounds__(256) void k_solve(
    const float* __restrict__ aff,      // may be nullptr -> recompute inline
    const float* __restrict__ emb,
    const float* __restrict__ pd,
    const float* __restrict__ weights,
    const float* __restrict__ pix,
    const float* __restrict__ Tmat,
    float* __restrict__ out)
{
    int blk = blockIdx.x;
    int b = blk / N_, i = blk % N_;
    int tid = threadIdx.x;
    float fx = pix[b*16+0], fy = pix[b*16+5], x0 = pix[b*16+2], y0 = pix[b*16+6];
    const float* Ti = Tmat + (size_t)(b*N_ + i)*16;
    float R00=Ti[0],R01=Ti[1],R02=Ti[2],  t0=Ti[3];
    float R10=Ti[4],R11=Ti[5],R12=Ti[6],  t1=Ti[7];
    float R20=Ti[8],R21=Ti[9],R22=Ti[10], t2=Ti[11];

    float ei[C_];
    float si = 0.f;
    if (!aff) {
        #pragma unroll
        for (int c = 0; c < C_; ++c) ei[c] = emb[(b*C_ + c)*N_ + i];
        si = 0.f;
    }

    float acc[27];
    #pragma unroll
    for (int j = 0; j < 27; ++j) acc[j] = 0.f;

    const float* arow = aff ? aff + ((size_t)b*N_ + i)*N_ : nullptr;

    for (int k = tid; k < N_; k += 256) {
        const float* p = pd + (size_t)(b*N_ + k)*8;
        float x = p[0], y = p[1], z = p[2];
        float g0 = p[3], g1 = p[4], g2 = p[5];
        float a;
        if (arow) {
            a = arow[k];
        } else {
            float d2s = 0.f;
            #pragma unroll
            for (int c = 0; c < C_; ++c) {
                float dd = ei[c] - emb[(b*C_ + c)*N_ + k];
                d2s += dd*dd;
            }
            a = expf(-sqrtf(d2s));
        }
        float w0 = a*weights[(b*3+0)*N_+k];
        float w1 = a*weights[(b*3+1)*N_+k];
        float w2 = a*weights[(b*3+2)*N_+k];
        float Px = R00*x + R01*y + R02*z + t0;
        float Py = R10*x + R11*y + R12*z + t1;
        float Pz = R20*x + R21*y + R22*z + t2;
        float d  = 1.0f / Pz;
        float dd2 = d*d;
        float uu = fx*Px*d + x0;
        float vv = fy*Py*d + y0;
        float rc[3] = { uu - g0, vv - g1, d - g2 };
        float wc[3] = { w0, w1, w2 };
        // Jp rows; Jac row = [jp0 jp1 jp2 | jp·hatcols(xyz_k)]
        float jp[3][3] = {
            { fx*d, 0.f,  -fx*Px*dd2 },
            { 0.f,  fy*d, -fy*Py*dd2 },
            { 0.f,  0.f,  -dd2       } };
        #pragma unroll
        for (int c = 0; c < 3; ++c) {
            float a0 = jp[c][0], a1 = jp[c][1], a2 = jp[c][2];
            float J[6] = { a0, a1, a2,
                           a1*z - a2*y,
                           a2*x - a0*z,
                           a0*y - a1*x };
            int idx = 0;
            #pragma unroll
            for (int pq = 0; pq < 6; ++pq) {
                float wjp = wc[c]*J[pq];
                #pragma unroll
                for (int q = pq; q < 6; ++q) acc[idx++] += wjp*J[q];
            }
            #pragma unroll
            for (int pq = 0; pq < 6; ++pq) acc[21+pq] += J[pq]*rc[c];  // rhs unweighted (ref)
        }
    }

    // wave32 reduce then cross-wave via LDS
    #pragma unroll
    for (int off = 16; off > 0; off >>= 1) {
        #pragma unroll
        for (int j = 0; j < 27; ++j) acc[j] += __shfl_down(acc[j], off, 32);
    }
    __shared__ float part[8][27];
    int wv = tid >> 5, ln = tid & 31;
    if (ln == 0) {
        #pragma unroll
        for (int j = 0; j < 27; ++j) part[wv][j] = acc[j];
    }
    __syncthreads();
    if (tid == 0) {
        float tot[27];
        #pragma unroll
        for (int j = 0; j < 27; ++j) {
            float s = 0.f;
            #pragma unroll
            for (int w = 0; w < 8; ++w) s += part[w][j];
            tot[j] = s;
        }
        // assemble augmented [H | rhs]
        float M[6][7];
        int idx = 0;
        for (int p = 0; p < 6; ++p)
            for (int q = p; q < 6; ++q) { M[p][q] = tot[idx]; M[q][p] = tot[idx]; ++idx; }
        for (int p = 0; p < 6; ++p) M[p][6] = tot[21+p];
        // Gauss-Jordan with partial pivoting
        for (int col = 0; col < 6; ++col) {
            int piv = col; float best = fabsf(M[col][col]);
            for (int r = col+1; r < 6; ++r) {
                float av = fabsf(M[r][col]);
                if (av > best) { best = av; piv = r; }
            }
            if (piv != col)
                for (int q = 0; q < 7; ++q) { float tv = M[col][q]; M[col][q] = M[piv][q]; M[piv][q] = tv; }
            float inv = 1.0f / M[col][col];
            for (int r = 0; r < 6; ++r) {
                if (r == col) continue;
                float f = M[r][col] * inv;
                for (int q = col; q < 7; ++q) M[r][q] -= f * M[col][q];
            }
        }
        float dl[6];
        for (int p = 0; p < 6; ++p) dl[p] = M[p][6] / M[p][p];
        // expmap(dl)
        float vx=dl[0], vy=dl[1], vz=dl[2], wx=dl[3], wy=dl[4], wz=dl[5];
        float th2 = wx*wx + wy*wy + wz*wz;
        float th = sqrtf(th2);
        float A, Bc, Cc;
        if (th < 1e-4f) {
            A  = 1.f  - th2*(1.f/6.f);
            Bc = 0.5f - th2*(1.f/24.f);
            Cc = (1.f/6.f) - th2*(1.f/120.f);
        } else {
            float sn = sinf(th), cs = cosf(th);
            A = sn/th; Bc = (1.f-cs)/th2; Cc = (th-sn)/(th2*th);
        }
        float Wm[3][3] = {{0,-wz,wy},{wz,0,-wx},{-wy,wx,0}};
        float W2[3][3];
        for (int r = 0; r < 3; ++r)
            for (int c = 0; c < 3; ++c) {
                float s = 0.f;
                for (int m2 = 0; m2 < 3; ++m2) s += Wm[r][m2]*Wm[m2][c];
                W2[r][c] = s;
            }
        float dT[4][4];
        for (int r = 0; r < 3; ++r)
            for (int c = 0; c < 3; ++c) {
                float id = (r == c) ? 1.f : 0.f;
                dT[r][c] = id + A*Wm[r][c] + Bc*W2[r][c];
            }
        float Vd[3][3];
        for (int r = 0; r < 3; ++r)
            for (int c = 0; c < 3; ++c) {
                float id = (r == c) ? 1.f : 0.f;
                Vd[r][c] = id + Bc*Wm[r][c] + Cc*W2[r][c];
            }
        dT[0][3] = Vd[0][0]*vx + Vd[0][1]*vy + Vd[0][2]*vz;
        dT[1][3] = Vd[1][0]*vx + Vd[1][1]*vy + Vd[1][2]*vz;
        dT[2][3] = Vd[2][0]*vx + Vd[2][1]*vy + Vd[2][2]*vz;
        dT[3][0] = 0.f; dT[3][1] = 0.f; dT[3][2] = 0.f; dT[3][3] = 1.f;
        // out = dT @ Tmat_i
        float* o = out + ((size_t)b*N_ + i)*16;
        for (int r = 0; r < 4; ++r)
            for (int c = 0; c < 4; ++c) {
                float s = 0.f;
                for (int m2 = 0; m2 < 4; ++m2) s += dT[r][m2]*Ti[m2*4 + c];
                o[r*4 + c] = s;
            }
    }
}

extern "C" void kernel_launch(void* const* d_in, const int* in_sizes, int n_in,
                              void* d_out, int out_size, void* d_ws, size_t ws_size,
                              hipStream_t stream) {
    const float* emb   = (const float*)d_in[0];   // (B,C,H,W)
    const float* rev   = (const float*)d_in[1];   // (B,3,H,W)
    const float* wts   = (const float*)d_in[2];   // (B,3,H,W)
    const float* depth = (const float*)d_in[3];   // (B,1,H,W)
    const float* pix   = (const float*)d_in[4];   // (B,4,4)
    const float* Tmat  = (const float*)d_in[5];   // (B,H,W,4,4)
    float* out = (float*)d_out;

    const size_t affElems = (size_t)B_*N_*N_;     // 2M floats = 8 MB
    const size_t pdElems  = (size_t)B_*N_*8;      // 64 KB
    const size_t need     = (affElems + pdElems) * sizeof(float);

    float *aff, *pd;
    bool haveAff = (ws_size >= need);
    if (haveAff) { aff = (float*)d_ws; pd = aff + affElems; }
    else         { aff = nullptr;      pd = (float*)d_ws; } // pd needs only 64 KB

    k_point<<<(B_*N_ + 255)/256, 256, 0, stream>>>(emb, rev, depth, pix, Tmat, pd);
    if (haveAff) {
        int waves = B_*NT*NT;                      // 8192 waves, 1 tile each
        k_aff<<<(waves*32 + 255)/256, 256, 0, stream>>>(emb, pd, aff);
    }
    k_solve<<<B_*N_, 256, 0, stream>>>(aff, emb, pd, wts, pix, Tmat, out);
}
